// Model_32358283608578
// MI455X (gfx1250) — matrix operand order repaired and checked by reading, not compiled, not run on hardware
//
#include <hip/hip_runtime.h>
#include <math.h>

typedef __attribute__((ext_vector_type(16))) _Float16 v16h;
typedef __attribute__((ext_vector_type(8)))  float    v8f;

#define HID    256
#define GATE   1024      // 4*HID
#define BATCH  64
#define TSTEPS 1024
#define NSPK   1251
#define ROWS   (TSTEPS*BATCH)   // 65536

// ---------------------------------------------------------------------------
// WMMA fragment helpers (contiguous-K layout shared by A and B, wave32)
// ---------------------------------------------------------------------------

// A-matrix 16x32 f16, row-major source with row stride `lda` (halves).
// lane<16: M=lane, K = 0..15 contiguous (Vv: K2v,2v+1)
// lane>=16: M=lane-16, K = 16..31 contiguous.
__device__ __forceinline__ v16h load_a_frag(const _Float16* __restrict__ base,
                                            int lda, int kbase, int lane) {
  int m  = lane & 15;
  int hi = lane >> 4;
  const _Float16* p = base + m * lda + kbase + hi * 16;
  union { v16h h; unsigned int u[8]; } a;
#pragma unroll
  for (int v = 0; v < 8; ++v) {
    int koff = 2 * v;
    a.u[v] = *(const unsigned int*)(p + koff);   // two consecutive halves (K=..,+1)
  }
  return a.h;
}

// B-matrix 32x16 f16 where B[k][n] = W[colbase+n][k], W row-major [N, ldw].
// lane holds column n=lane&15; halves j=0..15 are K = (lane>>4)*16 + j,
// i.e. 16 contiguous halves (32 bytes) of W's row -> two b128 loads.
__device__ __forceinline__ v16h load_b_frag(const _Float16* __restrict__ w,
                                            int ldw, int colbase, int kbase, int lane) {
  int n  = lane & 15;
  int hi = lane >> 4;
  return *(const v16h*)(w + (size_t)(colbase + n) * ldw + kbase + hi * 16);
}

__device__ __forceinline__ float sigf(float x) { return 1.0f / (1.0f + __expf(-x)); }

// ---------------------------------------------------------------------------
// Packing kernels
// ---------------------------------------------------------------------------
__global__ void k_convert_pad(const float* __restrict__ src, _Float16* __restrict__ dst,
                              int rows, int srcK, int dstK) {
  int idx = blockIdx.x * blockDim.x + threadIdx.x;
  if (idx >= rows * dstK) return;
  int r = idx / dstK, k = idx - r * dstK;
  float v = (k < srcK) ? src[r * srcK + k] : 0.0f;
  dst[idx] = (_Float16)v;
}

__global__ void k_bias(const float* __restrict__ bi, const float* __restrict__ bh,
                       float* __restrict__ bc) {
  int g = blockIdx.x * blockDim.x + threadIdx.x;
  if (g < GATE) bc[g] = bi[g] + bh[g];
}

// x: [B, T, 40] f32 -> XA: [T*B, 64] f16 (zero padded), row = t*B + b
__global__ void k_pack_x(const float* __restrict__ x, _Float16* __restrict__ xa) {
  int idx = blockIdx.x * blockDim.x + threadIdx.x;
  if (idx >= ROWS * 64) return;
  int row = idx >> 6, k = idx & 63;
  int t = row >> 6, b = row & 63;   // B == 64
  float v = (k < 40) ? x[(size_t)(b * TSTEPS + t) * 40 + k] : 0.0f;
  xa[idx] = (_Float16)v;
}

// ---------------------------------------------------------------------------
// Input-projection GEMM: XG[r, g] = f16( Xin[r,:K] . Wih[g,:K] + bc[g] )
// block = 8 waves; wave = 16 rows x 64 cols; grid = (ROWS/128, GATE/64)
// ---------------------------------------------------------------------------
__global__ void __launch_bounds__(256)
k_gemm_xg(const _Float16* __restrict__ xin, int lda, int K,
          const _Float16* __restrict__ wih, const float* __restrict__ bc,
          _Float16* __restrict__ xg) {
  int lane = threadIdx.x & 31;
  int wave = threadIdx.x >> 5;
  int rowbase = (blockIdx.x * 8 + wave) * 16;
  int colbase = blockIdx.y * 64;

  v8f acc[4] = {};
  for (int kb = 0; kb < K; kb += 32) {
    v16h a = load_a_frag(xin + (size_t)rowbase * lda, lda, kb, lane);
#pragma unroll
    for (int nt = 0; nt < 4; ++nt) {
      v16h b = load_b_frag(wih, K, colbase + nt * 16, kb, lane);
      acc[nt] = __builtin_amdgcn_wmma_f32_16x16x32_f16(
          false, a, false, b, (short)0, acc[nt], false, false);
    }
  }
  int n = lane & 15, hi = lane >> 4;
#pragma unroll
  for (int nt = 0; nt < 4; ++nt) {
    int col = colbase + nt * 16 + n;
    float bias = bc[col];
#pragma unroll
    for (int r = 0; r < 8; ++r) {
      int row = rowbase + r + hi * 8;
      xg[(size_t)row * GATE + col] = (_Float16)(acc[nt][r] + bias);
    }
  }
}

// ---------------------------------------------------------------------------
// Persistent recurrent kernel: 4 blocks x 512 threads (16 waves).
// Block owns batch rows [bbase, bbase+16). Per timestep:
//   G[16x1024] = XG[t] + h_f16[16x256] @ Whh^T   (16 waves x 4 tiles, K=256)
//   then fused LSTM cell update in LDS.
// ---------------------------------------------------------------------------
__global__ void __launch_bounds__(512)
k_lstm_rec(const _Float16* __restrict__ xg,   // [T*B, GATE], bias already fused
           const _Float16* __restrict__ whh,  // [GATE, HID] f16
           _Float16* __restrict__ hout) {     // [T*B, HID]
  __shared__ _Float16 sh_h[16 * HID];    //  8 KB : h state (f16 A-operand)
  __shared__ _Float16 sh_g[16 * GATE];   // 32 KB : gate pre-activations
  __shared__ float    sh_c[16 * HID];    // 16 KB : c state (f32)

  int tid   = threadIdx.x;
  int lane  = tid & 31;
  int wave  = tid >> 5;                  // 0..15 -> cols [wave*64, wave*64+64)
  int bbase = blockIdx.x * 16;

  for (int i = tid; i < 16 * HID; i += 512) { sh_h[i] = (_Float16)0.0f; sh_c[i] = 0.0f; }
  __syncthreads();

  int n = lane & 15, hi = lane >> 4;
  int erow = tid >> 5;                   // elementwise: row 0..15
  int ejb  = (tid & 31) * 8;             // elementwise: 8 cols per thread

  for (int t = 0; t < TSTEPS; ++t) {
    const _Float16* xgt = xg + (size_t)(t * BATCH + bbase) * GATE;
    __builtin_prefetch(xgt + (size_t)BATCH * GATE, 0, 1);  // next timestep's gates

    v8f acc[4] = {};
#pragma unroll
    for (int kb = 0; kb < 8; ++kb) {
      v16h a = load_a_frag(sh_h, HID, kb * 32, lane);
#pragma unroll
      for (int nt = 0; nt < 4; ++nt) {
        v16h b = load_b_frag(whh, HID, wave * 64 + nt * 16, kb * 32, lane);
        acc[nt] = __builtin_amdgcn_wmma_f32_16x16x32_f16(
            false, a, false, b, (short)0, acc[nt], false, false);
      }
    }
    // add precomputed input projection, stash gate matrix to LDS
#pragma unroll
    for (int nt = 0; nt < 4; ++nt) {
      int col = wave * 64 + nt * 16 + n;
#pragma unroll
      for (int r = 0; r < 8; ++r) {
        int m = r + hi * 8;
        float v = acc[nt][r] + (float)xgt[m * GATE + col];
        sh_g[m * GATE + col] = (_Float16)v;
      }
    }
    __syncthreads();

    // fused LSTM cell update (torch gate order: i, f, g, o)
#pragma unroll
    for (int q = 0; q < 8; ++q) {
      int j  = ejb + q;
      float iv = (float)sh_g[erow * GATE + j];
      float fv = (float)sh_g[erow * GATE + 256 + j];
      float gv = (float)sh_g[erow * GATE + 512 + j];
      float ov = (float)sh_g[erow * GATE + 768 + j];
      float c  = sigf(fv) * sh_c[erow * HID + j] + sigf(iv) * tanhf(gv);
      float h  = sigf(ov) * tanhf(c);
      sh_c[erow * HID + j] = c;
      _Float16 hh = (_Float16)h;
      sh_h[erow * HID + j] = hh;
      hout[(size_t)(t * BATCH + bbase + erow) * HID + j] = hh;
    }
    __syncthreads();
  }
}

// ---------------------------------------------------------------------------
// Classifier + log_softmax for batch element 63, one block per timestep.
// ---------------------------------------------------------------------------
__global__ void __launch_bounds__(256)
k_classifier(const _Float16* __restrict__ h2,  // [T*B, HID]
             const float* __restrict__ wlin,   // [NSPK, HID]
             const float* __restrict__ blin,
             float* __restrict__ out) {        // [T, NSPK]
  __shared__ float sh_h[HID];
  __shared__ float sh_logit[NSPK];
  __shared__ float red[256];
  int t = blockIdx.x, tid = threadIdx.x;

  const _Float16* hp = h2 + (size_t)(t * BATCH + (BATCH - 1)) * HID;
  sh_h[tid] = (float)hp[tid];
  __syncthreads();

  for (int o = tid; o < NSPK; o += 256) {
    const float* w = wlin + (size_t)o * HID;
    float s = blin[o];
#pragma unroll 4
    for (int k = 0; k < HID; ++k) s += sh_h[k] * w[k];
    sh_logit[o] = s;
  }
  __syncthreads();

  float m = -INFINITY;
  for (int o = tid; o < NSPK; o += 256) m = fmaxf(m, sh_logit[o]);
  red[tid] = m; __syncthreads();
  for (int s = 128; s > 0; s >>= 1) {
    if (tid < s) red[tid] = fmaxf(red[tid], red[tid + s]);
    __syncthreads();
  }
  float mx = red[0]; __syncthreads();

  float sum = 0.0f;
  for (int o = tid; o < NSPK; o += 256) sum += __expf(sh_logit[o] - mx);
  red[tid] = sum; __syncthreads();
  for (int s = 128; s > 0; s >>= 1) {
    if (tid < s) red[tid] += red[tid + s];
    __syncthreads();
  }
  float lse = mx + __logf(red[0]);
  __syncthreads();

  for (int o = tid; o < NSPK; o += 256)
    out[(size_t)t * NSPK + o] = sh_logit[o] - lse;
}

// ---------------------------------------------------------------------------
extern "C" void kernel_launch(void* const* d_in, const int* in_sizes, int n_in,
                              void* d_out, int out_size, void* d_ws, size_t ws_size,
                              hipStream_t stream) {
  (void)in_sizes; (void)n_in; (void)out_size; (void)ws_size;
  const float* x = (const float*)d_in[0];
  const float* w_ih[3] = {(const float*)d_in[1], (const float*)d_in[5], (const float*)d_in[9]};
  const float* w_hh[3] = {(const float*)d_in[2], (const float*)d_in[6], (const float*)d_in[10]};
  const float* b_ih[3] = {(const float*)d_in[3], (const float*)d_in[7], (const float*)d_in[11]};
  const float* b_hh[3] = {(const float*)d_in[4], (const float*)d_in[8], (const float*)d_in[12]};
  const float* w_lin = (const float*)d_in[13];
  const float* b_lin = (const float*)d_in[14];

  char* ws = (char*)d_ws;
  _Float16* XA  = (_Float16*)(ws);                          //   8 MB  [ROWS, 64]
  _Float16* XHA = (_Float16*)(ws + ((size_t)8  << 20));     //  32 MB  [ROWS, 256]
  _Float16* XHB = (_Float16*)(ws + ((size_t)40 << 20));     //  32 MB  [ROWS, 256]
  _Float16* XG  = (_Float16*)(ws + ((size_t)72 << 20));     // 128 MB  [ROWS, 1024]
  size_t woff = (size_t)200 << 20;
  _Float16* WIH[3]; _Float16* WHH[3]; float* BC[3];
  for (int l = 0; l < 3; ++l) { WIH[l] = (_Float16*)(ws + woff); woff += (size_t)512 << 10; }
  for (int l = 0; l < 3; ++l) { WHH[l] = (_Float16*)(ws + woff); woff += (size_t)512 << 10; }
  for (int l = 0; l < 3; ++l) { BC[l]  = (float*)(ws + woff);    woff += (size_t)16  << 10; }

  // --- pack weights / biases / input ---
  int kin[3] = {40, HID, HID}, kpad[3] = {64, HID, HID};
  for (int l = 0; l < 3; ++l) {
    int n = GATE * kpad[l];
    k_convert_pad<<<(n + 255) / 256, 256, 0, stream>>>(w_ih[l], WIH[l], GATE, kin[l], kpad[l]);
    k_convert_pad<<<(GATE * HID + 255) / 256, 256, 0, stream>>>(w_hh[l], WHH[l], GATE, HID, HID);
    k_bias<<<GATE / 256, 256, 0, stream>>>(b_ih[l], b_hh[l], BC[l]);
  }
  k_pack_x<<<(ROWS * 64) / 256, 256, 0, stream>>>(x, XA);

  // --- 3 stacked LSTM layers ---
  const _Float16* xin[3]  = {XA, XHA, XHB};
  int             lda[3]  = {64, HID, HID};
  _Float16*       hout[3] = {XHA, XHB, XHA};
  for (int l = 0; l < 3; ++l) {
    k_gemm_xg<<<dim3(ROWS / 128, GATE / 64), 256, 0, stream>>>(
        xin[l], lda[l], kpad[l], WIH[l], BC[l], XG);
    k_lstm_rec<<<4, 512, 0, stream>>>(XG, WHH[l], hout[l]);
  }

  // --- classifier + log_softmax on batch element 63 ---
  k_classifier<<<TSTEPS, 256, 0, stream>>>(XHA, w_lin, b_lin, (float*)d_out);
}
